// CDEDiscriminator_1864015806705
// MI455X (gfx1250) — compile-verified, matrix operand order checked by reading, not run on hardware
//
#include <hip/hip_runtime.h>

// ---------------------------------------------------------------------------
// Neural CDE forward for MI455X (gfx1250, wave32, WMMA 16x16x32 f16/f32-acc)
//
// B=2048, STEPS=128, OUT_DIM=32, HID=128.
// 64 workgroups x 256 threads (8 waves). Each WG owns 32 batch rows:
//   2 row-groups (16 rows, WMMA M=16) x 4 col-groups (waves) splitting the N
//   dimension of every GEMM. The 127-step RK scan runs entirely inside the
//   workgroup (batch rows are independent -> zero global sync).
// The bho,bo->bh contraction is done o-major: W3 is stored [o][h][k] so the
// contraction index o is a uniform loop and k accumulates in registers
// (no cross-lane shuffles, no predicated scatter stores).
// LDS ~128.5KB/WG -> 2 WGs (16 waves) per WGP for latency hiding.
// ---------------------------------------------------------------------------

typedef _Float16 h16;
typedef __attribute__((ext_vector_type(16))) _Float16 v16h;
typedef __attribute__((ext_vector_type(8)))  _Float16 v8h;
typedef __attribute__((ext_vector_type(4)))  _Float16 v4h;
typedef __attribute__((ext_vector_type(8)))  float    v8f;

#define B_TOT   2048
#define N_STEPS 128
#define OUT_D   32
#define HID_D   128

// ---- LDS layout sizes (bytes) ----
#define HEAD_BYTES  (4096*4 /*b3t*/ + 128*4 /*b1*/ + 128*4 /*w1row0*/ + 128*4 /*b2*/)
#define RG_BYTES    (16*128*4 /*h*/ + 4*16*128*4 /*k1..k4*/ \
                     + 16*128*2*3 /*A,z1,z2 f16*/ + 16*32*4 /*dx*/)
#define SMEM_BYTES  (HEAD_BYTES + 2*RG_BYTES)

__device__ __forceinline__ float lipswish_f(float x) {
  // 0.909 * x * sigmoid(x)
  float s = __builtin_amdgcn_rcpf(1.0f + __expf(-x));
  return 0.909f * x * s;
}

__device__ __forceinline__ float tanh_f(float x) {
  x = fminf(fmaxf(x, -15.0f), 15.0f);
  float e = __expf(-2.0f * x);
  return (1.0f - e) * __builtin_amdgcn_rcpf(1.0f + e);
}

// Load one 16x32 f16 WMMA A/B fragment for this lane from a row-major
// [row][128] matrix. rowbase = base + row*128 (row = M for A, N-col for B).
// Lanes 0-15 hold K = kc*32 + {0..7, 16..23},
// lanes 16-31 hold K = kc*32 + {8..15, 24..31}.
__device__ __forceinline__ v16h load_frag(const h16* __restrict__ rowbase,
                                          int kc, int halfsel) {
  int k0 = kc * 32 + (halfsel ? 8 : 0);
  v8h lo = *(const v8h*)(rowbase + k0);
  v8h hi = *(const v8h*)(rowbase + k0 + 16);
  return __builtin_shufflevector(lo, hi, 0,1,2,3,4,5,6,7,8,9,10,11,12,13,14,15);
}

__device__ __forceinline__ v8f wmma_f16(v16h a, v16h b, v8f c) {
  return __builtin_amdgcn_wmma_f32_16x16x32_f16(false, a, false, b,
                                                (short)0, c, false, false);
}

// ---------------------------------------------------------------------------
// Prep: transpose + f16-convert weights into workspace.
//   W1t[n][k]    = W1[k+1][n]      (129x128 -> 128x128, t-row split out)
//   W2t[n][k]    = W2[k][n]
//   W3g[o][h][k] = W3[k][h*32+o]   (o-major for in-register contraction)
// ---------------------------------------------------------------------------
__global__ void cde_prep(const float* __restrict__ W1, const float* __restrict__ W2,
                         const float* __restrict__ W3,
                         h16* __restrict__ W1t, h16* __restrict__ W2t,
                         h16* __restrict__ W3g, float* __restrict__ w1r0) {
  int i = blockIdx.x * blockDim.x + threadIdx.x;
  if (i < 128 * 128) {
    int n = i >> 7, k = i & 127;
    W1t[n * 128 + k] = (h16)W1[(k + 1) * 128 + n];
    W2t[n * 128 + k] = (h16)W2[k * 128 + n];
  }
  if (i < 128) w1r0[i] = W1[i];  // W1 row 0 (time channel), kept f32
  int nth = gridDim.x * blockDim.x;
  for (int e = i; e < 4096 * 128; e += nth) {
    int k = e & 127;
    int hcol = (e >> 7) & 127;
    int o = e >> 14;
    W3g[e] = (h16)W3[(size_t)k * 4096 + hcol * 32 + o];
  }
}

// ---------------------------------------------------------------------------
// Main persistent CDE kernel.
// ---------------------------------------------------------------------------
__global__ __launch_bounds__(256, 4)
void cde_main(const float* __restrict__ x, const float* __restrict__ times,
              const float* __restrict__ b1, const float* __restrict__ b2,
              const float* __restrict__ b3g,
              const float* __restrict__ Hw1, const float* __restrict__ Hb1,
              const float* __restrict__ Hw2, const float* __restrict__ Hb2,
              const float* __restrict__ Hw3, const float* __restrict__ Hb3,
              const float* __restrict__ Rw, const float* __restrict__ Rb,
              const h16* __restrict__ W1t, const h16* __restrict__ W2t,
              const h16* __restrict__ W3g, const float* __restrict__ w1r0,
              float* __restrict__ out) {
  extern __shared__ char smem_raw[];
  float* sB3  = (float*)smem_raw;                  // b3 transposed: [o][h], 4096
  float* sB1  = sB3 + 4096;
  float* sR0  = sB1 + 128;                         // W1 time row
  float* sB2  = sR0 + 128;
  char*  prg  = (char*)(sB2 + 128);

  const int tid  = threadIdx.x;
  const int lane = tid & 31;
  const int w    = tid >> 5;
  const int rg   = w >> 2;          // row-group 0..1
  const int cg   = w & 3;           // col-group 0..3
  const int rt   = tid & 127;       // thread index within row-group
  const int lrow = lane & 15;       // fragment row/col within tile
  const int hsel = lane >> 4;       // K-half select for fragments
  const int mrow = hsel * 8;        // C-layout row base for this lane

  char* rb = prg + (size_t)rg * RG_BYTES;
  float* sH  = (float*)rb;                 // 16x128 f32 state
  float* sK0 = sH  + 16 * 128;
  float* sK1 = sK0 + 16 * 128;
  float* sK2 = sK1 + 16 * 128;
  float* sK3 = sK2 + 16 * 128;
  h16*   sA  = (h16*)(sK3 + 16 * 128);     // f16 A-matrix input (h-arg)
  h16*   sZ1 = sA  + 16 * 128;
  h16*   sZ2 = sZ1 + 16 * 128;
  float* sDX = (float*)(sZ2 + 16 * 128);   // 16x32

  const int row0 = blockIdx.x * 32 + rg * 16;   // first batch row of this rg

  // ---- stage biases into LDS (b3 transposed to [o][h]) ----
  for (int i = tid; i < 4096; i += 256) {
    int o = i >> 7, hcol = i & 127;
    sB3[i] = b3g[hcol * 32 + o];
  }
  if (tid < 128) { sB1[tid] = b1[tid]; sB2[tid] = b2[tid]; sR0[tid] = w1r0[tid]; }
  __syncthreads();

  // ---- initial encoder: h0 = (lipswish(lipswish(x0@Hw1)@Hw2))@Hw3 + Hb3 ----
  for (int e = rt; e < 2048; e += 128) {           // e1 -> sK0
    int m = e >> 7, n = e & 127;
    const float* xr = x + (size_t)(row0 + m) * (N_STEPS * OUT_D);
    float acc = Hb1[n];
    for (int o = 0; o < 32; ++o) acc += xr[o] * Hw1[o * 128 + n];
    sK0[e] = lipswish_f(acc);
  }
  __syncthreads();
  for (int e = rt; e < 2048; e += 128) {           // e2 -> sK1
    int m = e >> 7, n = e & 127;
    float acc = Hb2[n];
    for (int k = 0; k < 128; ++k) acc += sK0[m * 128 + k] * Hw2[k * 128 + n];
    sK1[e] = lipswish_f(acc);
  }
  __syncthreads();
  for (int e = rt; e < 2048; e += 128) {           // h0 -> sH, sA
    int m = e >> 7, n = e & 127;
    float acc = Hb3[n];
    for (int k = 0; k < 128; ++k) acc += sK1[m * 128 + k] * Hw3[k * 128 + n];
    sH[e] = acc;
    sA[e] = (h16)acc;
  }
  __syncthreads();

  // ---- RK scan over 127 steps ----
  for (int s = 0; s < N_STEPS - 1; ++s) {
    float t0 = times[s];
    float dt = times[s + 1] - t0;
    float rdt = 1.0f / dt;

    // dXdt for this step -> LDS (one float4 pair per thread, 128B-coalesced)
    {
      int m  = rt >> 3;
      int o0 = (rt & 7) << 2;
      const float* xb = x + (size_t)(row0 + m) * (N_STEPS * OUT_D) + s * 32 + o0;
      float4 xn = *(const float4*)(xb + 32);
      float4 xc = *(const float4*)(xb);
      float4 d;
      d.x = (xn.x - xc.x) * rdt;  d.y = (xn.y - xc.y) * rdt;
      d.z = (xn.z - xc.z) * rdt;  d.w = (xn.w - xc.w) * rdt;
      *(float4*)(sDX + rt * 4) = d;
    }
    __syncthreads();

    const float dt3 = dt * (1.0f / 3.0f);
    float tv0 = t0, tv1 = t0 + dt3, tv2 = t0 + 2.0f * dt3, tv3 = t0 + dt;

    for (int stage = 0; stage < 4; ++stage) {
      float tcur = (stage == 0) ? tv0 : (stage == 1) ? tv1 : (stage == 2) ? tv2 : tv3;
      float* sKst = (stage == 0) ? sK0 : (stage == 1) ? sK1 : (stage == 2) ? sK2 : sK3;

      v16h aF[4];

      // ---- z1 = lipswish(harg@W1[1:] + t*W1[0] + b1), this wave's 32 cols ----
      {
        const h16* abase = sA + lrow * 128;
        #pragma unroll
        for (int kc = 0; kc < 4; ++kc) aF[kc] = load_frag(abase, kc, hsel);
      }
      #pragma unroll
      for (int tile = 0; tile < 2; ++tile) {
        int ncol = cg * 32 + tile * 16 + lrow;
        const h16* bbase = W1t + (size_t)ncol * 128;    // L2-resident stream
        v8f acc = {};
        #pragma unroll
        for (int kc = 0; kc < 4; ++kc)
          acc = wmma_f16(aF[kc], load_frag(bbase, kc, hsel), acc);
        float cadd = sB1[ncol] + tcur * sR0[ncol];
        #pragma unroll
        for (int i = 0; i < 8; ++i)
          sZ1[(mrow + i) * 128 + ncol] = (h16)lipswish_f(acc[i] + cadd);
      }
      __syncthreads();

      // ---- z2 = lipswish(z1@W2 + b2) ----
      {
        const h16* abase = sZ1 + lrow * 128;
        #pragma unroll
        for (int kc = 0; kc < 4; ++kc) aF[kc] = load_frag(abase, kc, hsel);
      }
      #pragma unroll
      for (int tile = 0; tile < 2; ++tile) {
        int ncol = cg * 32 + tile * 16 + lrow;
        const h16* bbase = W2t + (size_t)ncol * 128;
        v8f acc = {};
        #pragma unroll
        for (int kc = 0; kc < 4; ++kc)
          acc = wmma_f16(aF[kc], load_frag(bbase, kc, hsel), acc);
        float cadd = sB2[ncol];
        #pragma unroll
        for (int i = 0; i < 8; ++i)
          sZ2[(mrow + i) * 128 + ncol] = (h16)lipswish_f(acc[i] + cadd);
      }
      __syncthreads();

      // ---- k[m,h] = sum_o tanh(z2@W3[:,h,o] + b3[h,o]) * dx[m,o] ----
      // o-major: contraction accumulates in registers, no cross-lane traffic.
      {
        const h16* abase = sZ2 + lrow * 128;
        #pragma unroll
        for (int kc = 0; kc < 4; ++kc) aF[kc] = load_frag(abase, kc, hsel);
      }
      float kacc0[8] = {}, kacc1[8] = {};
      const int h0 = cg * 32 + lrow;          // this lane's h column (tile 0)
      #pragma unroll 2
      for (int o = 0; o < 32; ++o) {
        const h16* bb0 = W3g + (size_t)(o * 128 + h0) * 128;
        const h16* bb1 = W3g + (size_t)(o * 128 + h0 + 16) * 128;
        // speculative prefetch of next o-block rows (global_prefetch_b8)
        __builtin_prefetch(bb0 + 128 * 128, 0, 0);
        __builtin_prefetch(bb1 + 128 * 128, 0, 0);
        float dxv[8];                          // dx[mrow+i][o], LDS broadcast
        #pragma unroll
        for (int i = 0; i < 8; ++i) dxv[i] = sDX[(mrow + i) * 32 + o];
        v8f p0 = {}, p1 = {};
        #pragma unroll
        for (int kc = 0; kc < 4; ++kc) {
          p0 = wmma_f16(aF[kc], load_frag(bb0, kc, hsel), p0);
          p1 = wmma_f16(aF[kc], load_frag(bb1, kc, hsel), p1);
        }
        float bc0 = sB3[o * 128 + h0], bc1 = sB3[o * 128 + h0 + 16];
        #pragma unroll
        for (int i = 0; i < 8; ++i) {
          kacc0[i] += tanh_f(p0[i] + bc0) * dxv[i];
          kacc1[i] += tanh_f(p1[i] + bc1) * dxv[i];
        }
      }
      #pragma unroll
      for (int i = 0; i < 8; ++i) {
        sKst[(mrow + i) * 128 + h0]      = kacc0[i];
        sKst[(mrow + i) * 128 + h0 + 16] = kacc1[i];
      }
      __syncthreads();

      // ---- RK combine: next h-arg (and final h on stage 3), vectorized ----
      #pragma unroll
      for (int q = 0; q < 4; ++q) {
        int e = rt * 16 + q * 4;
        float4 h4  = *(float4*)(sH  + e);
        float4 k1q = *(float4*)(sK0 + e);
        float4 k2q = {}, k3q = {}, k4q = {};
        if (stage >= 1) k2q = *(float4*)(sK1 + e);
        if (stage >= 2) k3q = *(float4*)(sK2 + e);
        if (stage == 3) k4q = *(float4*)(sK3 + e);
        float* hp  = (float*)&h4;
        float* k1p = (float*)&k1q;
        float* k2p = (float*)&k2q;
        float* k3p = (float*)&k3q;
        float* k4p = (float*)&k4q;
        v4h av;
        #pragma unroll
        for (int i = 0; i < 4; ++i) {
          float h = hp[i], a = k1p[i], nh;
          if (stage == 0)      nh = h + dt3 * a;
          else if (stage == 1) nh = h + dt * (k2p[i] - (1.0f / 3.0f) * a);
          else if (stage == 2) nh = h + dt * (a - k2p[i] + k3p[i]);
          else { nh = h + dt * 0.125f * (a + 3.0f * (k2p[i] + k3p[i]) + k4p[i]); hp[i] = nh; }
          av[i] = (h16)nh;
        }
        if (stage == 3) *(float4*)(sH + e) = h4;
        *(v4h*)(sA + e) = av;
      }
      __syncthreads();
    }
  }

  // ---- readout: out = hT @ Rw + Rb ----
  if (rt < 16) {
    float acc = Rb[0];
    for (int n = 0; n < 128; ++n) acc += sH[rt * 128 + n] * Rw[n];
    out[row0 + rt] = acc;
  }
}

extern "C" void kernel_launch(void* const* d_in, const int* in_sizes, int n_in,
                              void* d_out, int out_size, void* d_ws, size_t ws_size,
                              hipStream_t stream) {
  (void)in_sizes; (void)n_in; (void)out_size; (void)ws_size;
  const float* x   = (const float*)d_in[0];
  const float* tms = (const float*)d_in[1];
  const float* W1  = (const float*)d_in[2];
  const float* b1  = (const float*)d_in[3];
  const float* W2  = (const float*)d_in[4];
  const float* b2  = (const float*)d_in[5];
  const float* W3  = (const float*)d_in[6];
  const float* b3  = (const float*)d_in[7];
  const float* Hw1 = (const float*)d_in[8];
  const float* Hb1 = (const float*)d_in[9];
  const float* Hw2 = (const float*)d_in[10];
  const float* Hb2 = (const float*)d_in[11];
  const float* Hw3 = (const float*)d_in[12];
  const float* Hb3 = (const float*)d_in[13];
  const float* Rw  = (const float*)d_in[14];
  const float* Rb  = (const float*)d_in[15];
  float* out = (float*)d_out;

  h16*   W1t  = (h16*)d_ws;               // 128*128
  h16*   W2t  = W1t + 128 * 128;          // 128*128
  h16*   W3g  = W2t + 128 * 128;          // 32*128*128 (o-major)
  float* w1r0 = (float*)(W3g + 4096 * 128);

  cde_prep<<<2048, 256, 0, stream>>>(W1, W2, W3, W1t, W2t, W3g, w1r0);
  cde_main<<<B_TOT / 32, 256, SMEM_BYTES, stream>>>(
      x, tms, b1, b2, b3, Hw1, Hb1, Hw2, Hb2, Hw3, Hb3, Rw, Rb,
      W1t, W2t, W3g, w1r0, out);
}
